// PtrNet1_8306466750969
// MI455X (gfx1250) — compile-verified, matrix operand order checked by reading, not run on hardware
//
#include <hip/hip_runtime.h>
#include <hip/hip_bf16.h>

// ---------------------------------------------------------------------------
// PtrNet (pointer network) for MI455X / gfx1250.
// All GEMMs run through v_wmma_f32_16x16x32_bf16 (bf16 A/B, f32 accum).
// Per-wave tiles enlarged (32x16x4gates LSTM, 64x16 GEMM) so A/B fragments
// are reused in registers: more WMMAs per B128 load.
// ---------------------------------------------------------------------------

#define Bn 512
#define Cn 100
#define En 128
#define Hn 512

typedef __bf16 bf16;
typedef __attribute__((ext_vector_type(16))) bf16  v16bf;
typedef __attribute__((ext_vector_type(8)))  bf16  v8bf;
typedef __attribute__((ext_vector_type(8)))  float v8f;

__device__ __forceinline__ float sigmoidf(float x) {
  return 1.0f / (1.0f + __expf(-x));
}

// Load one lane's 16-element bf16 fragment slice: two contiguous 8-element
// (16-byte) runs -> two global_load_b128. `p` = row_base + k0 + khalf*8.
__device__ __forceinline__ v16bf load_frag(const bf16* p) {
  v8bf lo = *reinterpret_cast<const v8bf*>(p);
  v8bf hi = *reinterpret_cast<const v8bf*>(p + 16);
  return __builtin_shufflevector(lo, hi,
      0, 1, 2, 3, 4, 5, 6, 7, 8, 9, 10, 11, 12, 13, 14, 15);
}

// ---------------------------------------------------------------------------
// Fused LSTM step:  g = Xt @ Wi^T + Hprev @ Wh^T + bias ; pointwise LSTM.
// One wave owns a 32(batch) x 16(hidden) tile, accumulating all FOUR gate
// tiles (N offsets q*H): i/f/g/o for each (b,h) land in the same lane, and
// each gate's B fragment is reused across the two batch sub-tiles.
// Grid: (B/32)*(H/16) = 16*32 = 512 waves.
// ---------------------------------------------------------------------------
__global__ __launch_bounds__(256) void lstm_step_kernel(
    const bf16* __restrict__ Xt, int ldx, int K1,      // B x K1 (row stride ldx)
    const bf16* __restrict__ Hprev,                    // B x H
    const bf16* __restrict__ Wi,                       // 4H x K1 row-major
    const bf16* __restrict__ Wh,                       // 4H x H  row-major
    const float* __restrict__ bias,                    // 4H  (bi+bh)
    const float* __restrict__ Cprev,                   // B x H f32
    float* __restrict__ Cnew,                          // B x H f32
    bf16* __restrict__ Hnext,                          // B x H bf16 (ld = H)
    bf16* __restrict__ Hseq, long long hseq_ld)        // optional seq copy
{
  const int lane  = threadIdx.x & 31;
  const int wid   = (blockIdx.x * blockDim.x + threadIdx.x) >> 5;
  const int tileM = (wid / (Hn / 16)) * 32;
  const int tileH = (wid % (Hn / 16)) * 16;
  const int khalf = lane >> 4;
  const int rA0   = tileM + (lane & 15);        // batch rows for the 2 A frags
  const int rA1   = rA0 + 16;
  const int rB    = tileH + (lane & 15);        // hidden col (= W row) for B

  v8f acc[2][4] = {};

  // Input contribution:  Xt @ Wi^T
  for (int k = 0; k < K1; k += 32) {
    v16bf a0 = load_frag(Xt + (size_t)rA0 * ldx + k + khalf * 8);
    v16bf a1 = load_frag(Xt + (size_t)rA1 * ldx + k + khalf * 8);
#pragma unroll
    for (int q = 0; q < 4; ++q) {
      v16bf b = load_frag(Wi + (size_t)(q * Hn + rB) * K1 + k + khalf * 8);
      acc[0][q] = __builtin_amdgcn_wmma_f32_16x16x32_bf16(
          false, a0, false, b, (short)0, acc[0][q], false, false);
      acc[1][q] = __builtin_amdgcn_wmma_f32_16x16x32_bf16(
          false, a1, false, b, (short)0, acc[1][q], false, false);
    }
  }
  // Recurrent contribution:  Hprev @ Wh^T
  for (int k = 0; k < Hn; k += 32) {
    v16bf a0 = load_frag(Hprev + (size_t)rA0 * Hn + k + khalf * 8);
    v16bf a1 = load_frag(Hprev + (size_t)rA1 * Hn + k + khalf * 8);
#pragma unroll
    for (int q = 0; q < 4; ++q) {
      v16bf b = load_frag(Wh + (size_t)(q * Hn + rB) * Hn + k + khalf * 8);
      acc[0][q] = __builtin_amdgcn_wmma_f32_16x16x32_bf16(
          false, a0, false, b, (short)0, acc[0][q], false, false);
      acc[1][q] = __builtin_amdgcn_wmma_f32_16x16x32_bf16(
          false, a1, false, b, (short)0, acc[1][q], false, false);
    }
  }

  // Pointwise LSTM epilogue. D layout: VGPR r -> M = base + r + 8*(lane>>4),
  // lane&15 -> N column. All four gates for (m,col) live in this lane.
  const int mhi = (lane >> 4) * 8;
  const int col = tileH + (lane & 15);
#pragma unroll
  for (int mt = 0; mt < 2; ++mt) {
#pragma unroll
    for (int r = 0; r < 8; ++r) {
      const int m = tileM + mt * 16 + r + mhi;
      float iv = acc[mt][0][r] + bias[0 * Hn + col];
      float fv = acc[mt][1][r] + bias[1 * Hn + col];
      float gv = acc[mt][2][r] + bias[2 * Hn + col];
      float ov = acc[mt][3][r] + bias[3 * Hn + col];
      float c  = sigmoidf(fv) * Cprev[(size_t)m * Hn + col] +
                 sigmoidf(iv) * tanhf(gv);
      float h  = sigmoidf(ov) * tanhf(c);
      Cnew[(size_t)m * Hn + col]  = c;
      bf16 hb = (bf16)h;
      Hnext[(size_t)m * Hn + col] = hb;
      if (Hseq) Hseq[(size_t)m * hseq_ld + col] = hb;
    }
  }
}

// ---------------------------------------------------------------------------
// Generic D(MxN) = A(MxK) @ W(NxK)^T, bf16 inputs, f32 output.
// One wave per 64x16 output tile: the B fragment is loaded once per K-step
// and reused by four WMMAs. Grid sized exactly by the host (M % 64 == 0).
// ---------------------------------------------------------------------------
__global__ __launch_bounds__(256) void gemm_bf16_wmma(
    const bf16* __restrict__ A, long long lda,
    const bf16* __restrict__ W, long long ldw,
    float* __restrict__ D, long long ldd,
    int Ntiles, int K)
{
  const int lane  = threadIdx.x & 31;
  const int wid   = (blockIdx.x * blockDim.x + threadIdx.x) >> 5;
  const int tileM = (wid / Ntiles) * 64;
  const int tileN = (wid % Ntiles) * 16;
  const int khalf = lane >> 4;
  const int rB    = tileN + (lane & 15);

  v8f acc[4] = {};
  for (int k = 0; k < K; k += 32) {
    v16bf b = load_frag(W + (size_t)rB * ldw + k + khalf * 8);
#pragma unroll
    for (int mt = 0; mt < 4; ++mt) {
      const int rA = tileM + mt * 16 + (lane & 15);
      v16bf a = load_frag(A + (size_t)rA * lda + k + khalf * 8);
      acc[mt] = __builtin_amdgcn_wmma_f32_16x16x32_bf16(
          false, a, false, b, (short)0, acc[mt], false, false);
    }
  }
  const int mhi = (lane >> 4) * 8;
  const int col = tileN + (lane & 15);
#pragma unroll
  for (int mt = 0; mt < 4; ++mt)
#pragma unroll
    for (int r = 0; r < 8; ++r)
      D[(size_t)(tileM + mt * 16 + r + mhi) * ldd + col] = acc[mt][r];
}

// ---------------------------------------------------------------------------
// Attention / pointer step. One 256-thread block per batch row.
//   u[c]  = CLIP*tanh(u1[b,c,:] + u2[b,:]) . v  -  INF*mask[b,c]
//   idx   = argmax(u);  nll += lse(u) - u[idx];  a = softmax(u)
//   din[b,:] = a @ emb[b]  (emitted directly as bf16 for the next LSTM step)
// ---------------------------------------------------------------------------
__global__ __launch_bounds__(256) void attention_kernel(
    int t,
    const float* __restrict__ u1,     // B x C x H
    const float* __restrict__ u2,     // B x H
    const float* __restrict__ v,      // H
    const float* __restrict__ embf,   // B x C x E (f32)
    float* __restrict__ mask,         // B x C
    float* __restrict__ nll,          // B
    bf16* __restrict__ din,           // B x E (next decoder input)
    float* __restrict__ out_nll,      // B (tail of d_out)
    int* __restrict__ out_idx)        // B x C (head of d_out)
{
  __shared__ float u_s[Cn];
  __shared__ float a_s[Cn];
  __shared__ float lse_s;

  const int b    = blockIdx.x;
  const int lane = threadIdx.x & 31;
  const int wave = threadIdx.x >> 5;

  for (int c = wave; c < Cn; c += 8) {
    const float* u1row = u1 + ((size_t)b * Cn + c) * Hn;
    const float* u2row = u2 + (size_t)b * Hn;
    float s = 0.f;
    for (int h = lane; h < Hn; h += 32)
      s += 10.0f * tanhf(u1row[h] + u2row[h]) * v[h];
    for (int o = 16; o; o >>= 1) s += __shfl_xor(s, o, 32);
    if (lane == 0) u_s[c] = s - 1.0e7f * mask[(size_t)b * Cn + c];
  }
  __syncthreads();

  if (threadIdx.x == 0) {
    float mx = u_s[0]; int idx = 0;
    for (int c = 1; c < Cn; ++c)
      if (u_s[c] > mx) { mx = u_s[c]; idx = c; }
    float se = 0.f;
    for (int c = 0; c < Cn; ++c) se += __expf(u_s[c] - mx);
    float lse = mx + __logf(se);
    lse_s = lse;
    float n = nll[b] + (lse - u_s[idx]);
    nll[b] = n;
    out_nll[b] = n;
    out_idx[(size_t)b * Cn + t] = idx;
    mask[(size_t)b * Cn + idx] += 1.0f;
  }
  __syncthreads();

  for (int c = threadIdx.x; c < Cn; c += blockDim.x)
    a_s[c] = __expf(u_s[c] - lse_s);
  __syncthreads();

  for (int e = threadIdx.x; e < En; e += blockDim.x) {
    float s = 0.f;
    const float* er = embf + (size_t)b * Cn * En + e;
    for (int c = 0; c < Cn; ++c) s += a_s[c] * er[(size_t)c * En];
    din[(size_t)b * En + e] = (bf16)s;
  }
}

// ---------------------------------------------------------------------------
// Small prep kernels.
// ---------------------------------------------------------------------------
__global__ void cvt_f32_bf16(const float* __restrict__ s, bf16* __restrict__ d,
                             int n) {
  int i = blockIdx.x * blockDim.x + threadIdx.x;
  if (i < n) d[i] = (bf16)s[i];
}

__global__ void add_f32(const float* __restrict__ a, const float* __restrict__ b,
                        float* __restrict__ o, int n) {
  int i = blockIdx.x * blockDim.x + threadIdx.x;
  if (i < n) o[i] = a[i] + b[i];
}

// emb[b,c,e] = x[b,c,0]*W_emb[e,0] + x[b,c,1]*W_emb[e,1]; f32 + bf16 copies.
__global__ void emb_kernel(const float* __restrict__ x,
                           const float* __restrict__ W_emb,
                           float* __restrict__ ef, bf16* __restrict__ eb,
                           int n) {
  int i = blockIdx.x * blockDim.x + threadIdx.x;
  if (i >= n) return;
  int e  = i & (En - 1);
  int bc = i >> 7;  // En == 128
  float r = x[(size_t)bc * 2 + 0] * W_emb[(size_t)e * 2 + 0] +
            x[(size_t)bc * 2 + 1] * W_emb[(size_t)e * 2 + 1];
  ef[i] = r;
  eb[i] = (bf16)r;
}

// ---------------------------------------------------------------------------
// Host orchestration.
// ---------------------------------------------------------------------------
extern "C" void kernel_launch(void* const* d_in, const int* in_sizes, int n_in,
                              void* d_out, int out_size, void* d_ws,
                              size_t ws_size, hipStream_t stream) {
  (void)in_sizes; (void)n_in; (void)out_size; (void)ws_size;

  const float* x      = (const float*)d_in[0];
  const float* dec0   = (const float*)d_in[1];
  const float* W_emb  = (const float*)d_in[2];
  const float* enc_Wi = (const float*)d_in[3];
  const float* enc_Wh = (const float*)d_in[4];
  const float* enc_bi = (const float*)d_in[5];
  const float* enc_bh = (const float*)d_in[6];
  const float* dec_Wi = (const float*)d_in[7];
  const float* dec_Wh = (const float*)d_in[8];
  const float* dec_bi = (const float*)d_in[9];
  const float* dec_bh = (const float*)d_in[10];
  const float* W_q    = (const float*)d_in[11];
  const float* W_ref  = (const float*)d_in[12];
  const float* vec    = (const float*)d_in[13];

  char* ws = (char*)d_ws;
  size_t off = 0;
  auto alloc = [&](size_t bytes) -> void* {
    void* p = ws + off;
    off = (off + bytes + 255) & ~(size_t)255;
    return p;
  };

  bf16*  encWi_b = (bf16*)alloc((size_t)4 * Hn * En * 2);
  bf16*  encWh_b = (bf16*)alloc((size_t)4 * Hn * Hn * 2);
  bf16*  decWi_b = (bf16*)alloc((size_t)4 * Hn * En * 2);
  bf16*  decWh_b = (bf16*)alloc((size_t)4 * Hn * Hn * 2);
  bf16*  Wq_b    = (bf16*)alloc((size_t)Hn * Hn * 2);
  bf16*  Wref_b  = (bf16*)alloc((size_t)Hn * Hn * 2);
  float* b_enc   = (float*)alloc((size_t)4 * Hn * 4);
  float* b_dec   = (float*)alloc((size_t)4 * Hn * 4);
  float* embf    = (float*)alloc((size_t)Bn * Cn * En * 4);
  bf16*  embb    = (bf16*)alloc((size_t)Bn * Cn * En * 2);
  bf16*  enc_hb  = (bf16*)alloc((size_t)Bn * Cn * Hn * 2);
  float* u1      = (float*)alloc((size_t)Bn * Cn * Hn * 4);
  float* u2      = (float*)alloc((size_t)Bn * Hn * 4);
  bf16*  hb[2]   = {(bf16*)alloc((size_t)Bn * Hn * 2),
                    (bf16*)alloc((size_t)Bn * Hn * 2)};
  float* cb[2]   = {(float*)alloc((size_t)Bn * Hn * 4),
                    (float*)alloc((size_t)Bn * Hn * 4)};
  bf16*  din_b   = (bf16*)alloc((size_t)Bn * En * 2);
  float* mask    = (float*)alloc((size_t)Bn * Cn * 4);
  float* nll     = (float*)alloc((size_t)Bn * 4);

  int*   out_idx = (int*)d_out;                      // pred_tour (B*C int32)
  float* out_nll = (float*)d_out + (size_t)Bn * Cn;  // neg_log (B f32)

  const int TPB = 256;
  auto nb = [](long long n) { return (int)((n + 255) / 256); };

  // --- prep: bf16 weights, fused biases, embeddings, zeroed state ---------
  cvt_f32_bf16<<<nb(4 * Hn * En), TPB, 0, stream>>>(enc_Wi, encWi_b, 4 * Hn * En);
  cvt_f32_bf16<<<nb(4 * Hn * Hn), TPB, 0, stream>>>(enc_Wh, encWh_b, 4 * Hn * Hn);
  cvt_f32_bf16<<<nb(4 * Hn * En), TPB, 0, stream>>>(dec_Wi, decWi_b, 4 * Hn * En);
  cvt_f32_bf16<<<nb(4 * Hn * Hn), TPB, 0, stream>>>(dec_Wh, decWh_b, 4 * Hn * Hn);
  cvt_f32_bf16<<<nb(Hn * Hn), TPB, 0, stream>>>(W_q, Wq_b, Hn * Hn);
  cvt_f32_bf16<<<nb(Hn * Hn), TPB, 0, stream>>>(W_ref, Wref_b, Hn * Hn);
  add_f32<<<nb(4 * Hn), TPB, 0, stream>>>(enc_bi, enc_bh, b_enc, 4 * Hn);
  add_f32<<<nb(4 * Hn), TPB, 0, stream>>>(dec_bi, dec_bh, b_dec, 4 * Hn);
  emb_kernel<<<nb((long long)Bn * Cn * En), TPB, 0, stream>>>(
      x, W_emb, embf, embb, Bn * Cn * En);
  cvt_f32_bf16<<<nb(Bn * En), TPB, 0, stream>>>(dec0, din_b, Bn * En);

  hipMemsetAsync(hb[0], 0, (size_t)Bn * Hn * 2, stream);
  hipMemsetAsync(cb[0], 0, (size_t)Bn * Hn * 4, stream);
  hipMemsetAsync(mask, 0, (size_t)Bn * Cn * 4, stream);
  hipMemsetAsync(nll, 0, (size_t)Bn * 4, stream);

  const int lstm_waves  = (Bn / 32) * (Hn / 16);          // 512
  const int lstm_blocks = lstm_waves * 32 / TPB;          // 64

  // --- encoder: 100 fused WMMA-LSTM steps ---------------------------------
  for (int t = 0; t < Cn; ++t) {
    lstm_step_kernel<<<lstm_blocks, TPB, 0, stream>>>(
        embb + (size_t)t * En, Cn * En, En,
        hb[t & 1], encWi_b, encWh_b, b_enc,
        cb[t & 1], cb[(t + 1) & 1], hb[(t + 1) & 1],
        enc_hb + (size_t)t * Hn, (long long)Cn * Hn);
  }

  // --- u1 = enc_h @ W_ref^T : one big WMMA GEMM (51200 x 512 x 512) -------
  {
    long long waves = (long long)(Bn * Cn / 64) * (Hn / 16);  // 25600
    gemm_bf16_wmma<<<(int)(waves * 32 / TPB), TPB, 0, stream>>>(
        enc_hb, Hn, Wref_b, Hn, u1, Hn, Hn / 16, Hn);
  }

  // --- decoder: LSTM + u2 GEMM + attention per step -----------------------
  for (int t = 0; t < Cn; ++t) {
    lstm_step_kernel<<<lstm_blocks, TPB, 0, stream>>>(
        din_b, En, En,
        hb[t & 1], decWi_b, decWh_b, b_dec,
        cb[t & 1], cb[(t + 1) & 1], hb[(t + 1) & 1],
        (bf16*)nullptr, 0);

    long long waves = (long long)(Bn / 64) * (Hn / 16);       // 256
    gemm_bf16_wmma<<<(int)(waves * 32 / TPB), TPB, 0, stream>>>(
        hb[(t + 1) & 1], Hn, Wq_b, Hn, u2, Hn, Hn / 16, Hn);

    attention_kernel<<<Bn, TPB, 0, stream>>>(
        t, u1, u2, vec, embf, mask, nll, din_b, out_nll, out_idx);
  }
}
